// RBTBlock_dual_5566277616596
// MI455X (gfx1250) — matrix operand order repaired and checked by reading, not compiled, not run on hardware
//
#include <hip/hip_runtime.h>

// ---------------------------------------------------------------------------
// MI455X (gfx1250) fused RBTBlock. Key optimization: the two bias-only MLPs
// collapse to single affine maps (no activation between layers), so the whole
// problem becomes memory-bound (~200MB traffic -> ~9us at 23.3 TB/s). All
// GEMMs use v_wmma_f32_16x16x32_bf16 (fp32 accumulate).
// ---------------------------------------------------------------------------

typedef __attribute__((ext_vector_type(16))) __bf16          v16bf;
typedef __attribute__((ext_vector_type(8)))  float           v8f;
typedef __attribute__((ext_vector_type(8)))  unsigned short  v8us;

union FragAB {
    v16bf v;
    v8us  h[2];
    unsigned short s[16];
};

static __device__ __forceinline__ unsigned short f2bf(float f) {
    unsigned u = __float_as_uint(f);
    unsigned r = u + 0x7FFFu + ((u >> 16) & 1u);   // round-to-nearest-even
    return (unsigned short)(r >> 16);
}

static __device__ __forceinline__ v8f vzero8() {
    v8f z = {0.f, 0.f, 0.f, 0.f, 0.f, 0.f, 0.f, 0.f};
    return z;
}

static __device__ __forceinline__ v8f wmma_bf16(const FragAB& a, const FragAB& b, v8f c) {
    return __builtin_amdgcn_wmma_f32_16x16x32_bf16(false, a.v, false, b.v,
                                                   (short)0, c, false, false);
}

constexpr int   NB     = 8192;     // batch
constexpr int   ND     = 2048;     // feature dim
constexpr float EPS_BN = 2e-5f;

// ---------------------------- conversion kernels ---------------------------

__global__ void cvt_kernel(const float* __restrict__ src, unsigned short* __restrict__ dst, int n) {
    int i = blockIdx.x * blockDim.x + threadIdx.x;
    if (i < n) dst[i] = f2bf(src[i]);
}

__global__ void zero_kernel(unsigned short* __restrict__ dst, int n) {
    int i = blockIdx.x * blockDim.x + threadIdx.x;
    if (i < n) dst[i] = 0;
}

// src [R,C] fp32 row-major -> dst [C,R] bf16 row-major
__global__ void cvtT_kernel(const float* __restrict__ src, unsigned short* __restrict__ dst,
                            int R, int C) {
    int i = blockIdx.x * blockDim.x + threadIdx.x;
    if (i >= R * C) return;
    int r = i / C, c = i - r * C;
    dst[(size_t)c * R + r] = f2bf(src[i]);
}

// prototype [16, ND] -> protoTb [ND, 32] bf16, k >= 16 zero padded
__global__ void protoT_kernel(const float* __restrict__ proto, unsigned short* __restrict__ dst) {
    int i = blockIdx.x * blockDim.x + threadIdx.x;
    if (i >= ND * 32) return;
    int d = i >> 5, k = i & 31;
    dst[i] = (k < 16) ? f2bf(proto[(size_t)k * ND + d]) : (unsigned short)0;
}

// -------------------------- bias composition -------------------------------
// beff_c = cW3 (cW2 cb1 + cb2) + cb3 (16), beff_a similarly (2). beff[18].
__global__ void bias_kernel(const float* __restrict__ cW2, const float* __restrict__ cb1,
                            const float* __restrict__ cb2, const float* __restrict__ cW3,
                            const float* __restrict__ cb3, const float* __restrict__ aW2,
                            const float* __restrict__ ab1, const float* __restrict__ ab2,
                            const float* __restrict__ aW3, const float* __restrict__ ab3,
                            float* __restrict__ beff) {
    __shared__ float bv[1024];
    int t = threadIdx.x;
    if (t < 512) {
        float s = 0.f;
        const float* w = cW2 + (size_t)t * 1024;
        for (int k = 0; k < 1024; ++k) s += w[k] * cb1[k];
        bv[t] = s + cb2[t];
    } else {
        int u = t - 512;
        float s = 0.f;
        const float* w = aW2 + (size_t)u * 1024;
        for (int k = 0; k < 1024; ++k) s += w[k] * ab1[k];
        bv[t] = s + ab2[u];
    }
    __syncthreads();
    if (t < 16) {
        float s = 0.f;
        const float* w = cW3 + (size_t)t * 512;
        for (int k = 0; k < 512; ++k) s += w[k] * bv[k];
        beff[t] = s + cb3[t];
    } else if (t < 18) {
        int u = t - 16;
        float s = 0.f;
        const float* w = aW3 + (size_t)u * 512;
        for (int k = 0; k < 512; ++k) s += w[k] * bv[512 + k];
        beff[16 + u] = s + ab3[u];
    }
}

// --------------------- generic bf16 WMMA GEMM (precompute) -----------------
// C[M,N] = A[M,K] @ W[N,K]^T ; mode 1: bf16 C[M,N]; mode 2: bf16 C^T [N,M].
// One wave per 16x16 tile, 4 waves/block. Wave-uniform early exit only.
__global__ void gemm_bf16_kernel(const unsigned short* __restrict__ A, int lda,
                                 const unsigned short* __restrict__ W, int ldw,
                                 int M, int N, int K,
                                 unsigned short* __restrict__ outB, int ldo, int mode) {
    int lane = threadIdx.x & 31;
    int wave = threadIdx.x >> 5;
    int tilesN = N >> 4;
    int tiles  = ((M + 15) >> 4) * tilesN;
    int idx = blockIdx.x * 4 + wave;
    if (idx >= tiles) return;                 // uniform per wave
    int tm = idx / tilesN, tn = idx - tm * tilesN;
    int nlane = lane & 15, half = lane >> 4;
    int rowA = tm * 16 + nlane;
    if (rowA >= M) rowA = M - 1;              // clamp, garbage rows not stored
    const unsigned short* abase = A + (size_t)rowA * lda;
    const unsigned short* wbase = W + (size_t)(tn * 16 + nlane) * ldw;
    v8f acc = vzero8();
    for (int kb = 0; kb < K; kb += 32) {
        FragAB a, b;
        const unsigned short* ar = abase + kb + half * 8;
        a.h[0] = *(const v8us*)(ar);
        a.h[1] = *(const v8us*)(ar + 16);
        const unsigned short* wr = wbase + kb + half * 8;
        b.h[0] = *(const v8us*)(wr);
        b.h[1] = *(const v8us*)(wr + 16);
        acc = wmma_bf16(a, b, acc);
    }
#pragma unroll
    for (int r = 0; r < 8; ++r) {
        int gm = tm * 16 + r + half * 8;
        if (gm >= M) continue;
        int gn = tn * 16 + nlane;
        if (mode == 1) outB[(size_t)gm * ldo + gn] = f2bf(acc[r]);
        else           outB[(size_t)gn * ldo + gm] = f2bf(acc[r]);
    }
}

// ------------------------- G1: fused first layer ---------------------------
// For each 16-row stripe: C[16,160] = feat_bf16[16,2048] @ Wbig[160,2048]^T.
// Tiles 0..7 -> BN1+PReLU1 -> t1b bf16 [4][NB][32]; tiles 8,9 -> raw logits.
__global__ void g1_kernel(const float* __restrict__ feat,
                          const unsigned short* __restrict__ Wbig,
                          const float* __restrict__ pg1, const float* __restrict__ pb1,
                          const float* __restrict__ pm1, const float* __restrict__ pv1,
                          const float* __restrict__ pa1,
                          unsigned short* __restrict__ t1b,
                          float* __restrict__ logits) {
    int lane  = threadIdx.x & 31;
    int wave  = threadIdx.x >> 5;
    int rows0 = blockIdx.x * 128 + wave * 16;
    int nlane = lane & 15, half = lane >> 4;

    v8f acc[10];
#pragma unroll
    for (int t = 0; t < 10; ++t) acc[t] = vzero8();

    const float* arow = feat + (size_t)(rows0 + nlane) * ND;
    for (int kb = 0; kb < ND; kb += 32) {
        // A fragment: convert fp32 -> bf16 on the fly (feat read exactly once)
        const float* ar = arow + kb + half * 8;
        float4 f0 = *(const float4*)(ar);
        float4 f1 = *(const float4*)(ar + 4);
        float4 f2 = *(const float4*)(ar + 16);
        float4 f3 = *(const float4*)(ar + 20);
        FragAB a;
        a.s[0]  = f2bf(f0.x); a.s[1]  = f2bf(f0.y); a.s[2]  = f2bf(f0.z); a.s[3]  = f2bf(f0.w);
        a.s[4]  = f2bf(f1.x); a.s[5]  = f2bf(f1.y); a.s[6]  = f2bf(f1.z); a.s[7]  = f2bf(f1.w);
        a.s[8]  = f2bf(f2.x); a.s[9]  = f2bf(f2.y); a.s[10] = f2bf(f2.z); a.s[11] = f2bf(f2.w);
        a.s[12] = f2bf(f3.x); a.s[13] = f2bf(f3.y); a.s[14] = f2bf(f3.z); a.s[15] = f2bf(f3.w);
#pragma unroll
        for (int t = 0; t < 10; ++t) {
            FragAB b;
            const unsigned short* wr =
                Wbig + (size_t)(t * 16 + nlane) * ND + kb + half * 8;
            b.h[0] = *(const v8us*)(wr);
            b.h[1] = *(const v8us*)(wr + 16);
            acc[t] = wmma_bf16(a, b, acc[t]);
        }
    }
    // epilogue: BN1 + PReLU1 on channels 0..127
#pragma unroll
    for (int t = 0; t < 8; ++t) {
        int c = t * 16 + nlane;                 // global channel == Wbig row
        float s  = pg1[c] * rsqrtf(pv1[c] + EPS_BN);
        float tt = pb1[c] - pm1[c] * s;
        float al = pa1[c];
        int p = c >> 5, j = c & 31;
#pragma unroll
        for (int r = 0; r < 8; ++r) {
            float y = acc[t][r] * s + tt;
            y = y > 0.f ? y : al * y;
            int row = rows0 + r + half * 8;
            t1b[((size_t)p * NB + row) * 32 + j] = f2bf(y);
        }
    }
#pragma unroll
    for (int r = 0; r < 8; ++r) {
        int row = rows0 + r + half * 8;
        logits[(size_t)row * 18 + nlane] = acc[8][r];
        if (nlane < 2) logits[(size_t)row * 18 + 16 + nlane] = acc[9][r];
    }
}

// ------------------------------ softmaxes ----------------------------------
__global__ void softmax_kernel(const float* __restrict__ logits, const float* __restrict__ beff,
                               unsigned short* __restrict__ compab, float* __restrict__ alpha) {
    int b = blockIdx.x * blockDim.x + threadIdx.x;
    if (b >= NB) return;
    const float* l = logits + (size_t)b * 18;
    float v[16], mx = -3.4e38f;
#pragma unroll
    for (int j = 0; j < 16; ++j) { v[j] = l[j] + beff[j]; mx = fmaxf(mx, v[j]); }
    float s = 0.f;
#pragma unroll
    for (int j = 0; j < 16; ++j) { v[j] = __expf(v[j] - mx); s += v[j]; }
    float inv = 1.f / s;
#pragma unroll
    for (int j = 0; j < 16; ++j) compab[(size_t)b * 32 + j] = f2bf(v[j] * inv);
#pragma unroll
    for (int j = 16; j < 32; ++j) compab[(size_t)b * 32 + j] = 0;   // K-pad
    float l0 = l[16] + beff[16], l1 = l[17] + beff[17];
    float m2 = fmaxf(l0, l1);
    float e0 = __expf(l0 - m2), e1 = __expf(l1 - m2);
    float inv2 = 1.f / (e0 + e1);
    alpha[(size_t)b * 2 + 0] = e0 * inv2;
    alpha[(size_t)b * 2 + 1] = e1 * inv2;
}

// ------------------- G2: t2 = BN2PReLU2(t1 @ pW2^T) ------------------------
// block = 4 waves; wave w handles p=w for one 16-row b-tile, N=32 (2 tiles).
__global__ void g2_kernel(const unsigned short* __restrict__ t1b,
                          const unsigned short* __restrict__ pW2b,
                          const float* __restrict__ pg2, const float* __restrict__ pb2,
                          const float* __restrict__ pm2, const float* __restrict__ pv2,
                          const float* __restrict__ pa2,
                          unsigned short* __restrict__ t2b) {
    int lane = threadIdx.x & 31;
    int p    = threadIdx.x >> 5;
    int rows0 = blockIdx.x * 16;
    int nlane = lane & 15, half = lane >> 4;

    FragAB a;
    const unsigned short* ar = t1b + ((size_t)p * NB + rows0 + nlane) * 32 + half * 8;
    a.h[0] = *(const v8us*)(ar);
    a.h[1] = *(const v8us*)(ar + 16);
#pragma unroll
    for (int nt = 0; nt < 2; ++nt) {
        FragAB b;
        const unsigned short* wr = pW2b + ((size_t)p * 32 + nt * 16 + nlane) * 32 + half * 8;
        b.h[0] = *(const v8us*)(wr);
        b.h[1] = *(const v8us*)(wr + 16);
        v8f acc = wmma_bf16(a, b, vzero8());
        int j = nt * 16 + nlane;
        int c = p * 32 + j;
        float s  = pg2[c] * rsqrtf(pv2[c] + EPS_BN);
        float tt = pb2[c] - pm2[c] * s;
        float al = pa2[c];
#pragma unroll
        for (int r = 0; r < 8; ++r) {
            float y = acc[r] * s + tt;
            y = y > 0.f ? y : al * y;
            int row = rows0 + r + half * 8;
            t2b[((size_t)p * NB + row) * 32 + j] = f2bf(y);
        }
    }
}

// ----------------- final fused kernel (carries main traffic) ---------------
// out[b,d] = feat + a0 * sum_p BN3PReLU3(t2_p @ pW3_p^T) + a1 * (compa @ proto)
__global__ void final_kernel(const float* __restrict__ feat,
                             const unsigned short* __restrict__ t2b,
                             const unsigned short* __restrict__ pW3b,
                             const float* __restrict__ pg3, const float* __restrict__ pb3,
                             const float* __restrict__ pm3, const float* __restrict__ pv3,
                             const float* __restrict__ pa3,
                             const unsigned short* __restrict__ compab,
                             const unsigned short* __restrict__ protoTb,
                             const float* __restrict__ alpha,
                             float* __restrict__ out) {
    int lane  = threadIdx.x & 31;
    int wave  = threadIdx.x >> 5;
    int rows0 = blockIdx.y * 128 + wave * 16;
    int dbase = blockIdx.x * 16;
    int nlane = lane & 15, half = lane >> 4;

    // attention path: [16,32](K-padded) @ proto^T tile [16,32]
    FragAB ca, pf;
    const unsigned short* car = compab + (size_t)(rows0 + nlane) * 32 + half * 8;
    ca.h[0] = *(const v8us*)(car);
    ca.h[1] = *(const v8us*)(car + 16);
    const unsigned short* prr = protoTb + (size_t)(dbase + nlane) * 32 + half * 8;
    pf.h[0] = *(const v8us*)(prr);
    pf.h[1] = *(const v8us*)(prr + 16);
    v8f att = wmma_bf16(ca, pf, vzero8());

    // transform path: 4 x (K=32 WMMA) + BN3/PReLU3 + sum over p
    v8f sum = vzero8();
    int d = dbase + nlane;
#pragma unroll
    for (int p = 0; p < 4; ++p) {
        FragAB a, w;
        const unsigned short* ar = t2b + ((size_t)p * NB + rows0 + nlane) * 32 + half * 8;
        a.h[0] = *(const v8us*)(ar);
        a.h[1] = *(const v8us*)(ar + 16);
        const unsigned short* wr = pW3b + ((size_t)p * ND + dbase + nlane) * 32 + half * 8;
        w.h[0] = *(const v8us*)(wr);
        w.h[1] = *(const v8us*)(wr + 16);
        v8f acc = wmma_bf16(a, w, vzero8());
        int c = p * ND + d;
        float s  = pg3[c] * rsqrtf(pv3[c] + EPS_BN);
        float tt = pb3[c] - pm3[c] * s;
        float al = pa3[c];
#pragma unroll
        for (int r = 0; r < 8; ++r) {
            float y = acc[r] * s + tt;
            y = y > 0.f ? y : al * y;
            sum[r] += y;
        }
    }
#pragma unroll
    for (int r = 0; r < 8; ++r) {
        int row = rows0 + r + half * 8;
        float a0 = alpha[(size_t)row * 2 + 0];
        float a1 = alpha[(size_t)row * 2 + 1];
        size_t o = (size_t)row * ND + dbase + nlane;
        out[o] = feat[o] + a0 * sum[r] + a1 * att[r];
    }
}

// ---------------------------------------------------------------------------

extern "C" void kernel_launch(void* const* d_in, const int* in_sizes, int n_in,
                              void* d_out, int out_size, void* d_ws, size_t ws_size,
                              hipStream_t stream) {
    (void)in_sizes; (void)n_in; (void)out_size; (void)ws_size;

    const float* feat = (const float*)d_in[0];
    const float* pW1  = (const float*)d_in[1];
    const float* pg1  = (const float*)d_in[2];
    const float* pb1  = (const float*)d_in[3];
    const float* pm1  = (const float*)d_in[4];
    const float* pv1  = (const float*)d_in[5];
    const float* pa1  = (const float*)d_in[6];
    const float* pW2  = (const float*)d_in[7];
    const float* pg2  = (const float*)d_in[8];
    const float* pb2  = (const float*)d_in[9];
    const float* pm2  = (const float*)d_in[10];
    const float* pv2  = (const float*)d_in[11];
    const float* pa2  = (const float*)d_in[12];
    const float* pW3  = (const float*)d_in[13];
    const float* pg3  = (const float*)d_in[14];
    const float* pb3  = (const float*)d_in[15];
    const float* pm3  = (const float*)d_in[16];
    const float* pv3  = (const float*)d_in[17];
    const float* pa3  = (const float*)d_in[18];
    const float* cW1  = (const float*)d_in[19];
    const float* cb1  = (const float*)d_in[20];
    const float* cW2  = (const float*)d_in[21];
    const float* cb2  = (const float*)d_in[22];
    const float* cW3  = (const float*)d_in[23];
    const float* cb3  = (const float*)d_in[24];
    const float* aW1  = (const float*)d_in[25];
    const float* ab1  = (const float*)d_in[26];
    const float* aW2  = (const float*)d_in[27];
    const float* ab2  = (const float*)d_in[28];
    const float* aW3  = (const float*)d_in[29];
    const float* ab3  = (const float*)d_in[30];
    const float* prototype = (const float*)d_in[31];

    // ---- workspace bump allocator (total ~21.5 MB, 256B aligned) ----
    unsigned char* wp = (unsigned char*)d_ws;
    auto alloc = [&](size_t bytes) -> unsigned char* {
        unsigned char* r = wp;
        wp += (bytes + 255) & ~(size_t)255;
        return r;
    };
    unsigned short* Wbig    = (unsigned short*)alloc((size_t)160 * ND * 2);
    unsigned short* cW1Tb   = (unsigned short*)alloc((size_t)ND * 1024 * 2);
    unsigned short* aW1Tb   = (unsigned short*)alloc((size_t)ND * 1024 * 2);
    unsigned short* cW2b    = (unsigned short*)alloc((size_t)512 * 1024 * 2);
    unsigned short* aW2b    = (unsigned short*)alloc((size_t)512 * 1024 * 2);
    unsigned short* cW3b    = (unsigned short*)alloc((size_t)16 * 512 * 2);
    unsigned short* aW3b    = (unsigned short*)alloc((size_t)2 * 512 * 2);
    unsigned short* tmp1cT  = (unsigned short*)alloc((size_t)ND * 512 * 2);
    unsigned short* tmp1aT  = (unsigned short*)alloc((size_t)ND * 512 * 2);
    unsigned short* pW2b    = (unsigned short*)alloc((size_t)4 * 32 * 32 * 2);
    unsigned short* pW3b    = (unsigned short*)alloc((size_t)4 * ND * 32 * 2);
    unsigned short* protoTb = (unsigned short*)alloc((size_t)ND * 32 * 2);
    unsigned short* t1b     = (unsigned short*)alloc((size_t)4 * NB * 32 * 2);
    unsigned short* t2b     = (unsigned short*)alloc((size_t)4 * NB * 32 * 2);
    float*          logits  = (float*)alloc((size_t)NB * 18 * 4);
    unsigned short* compab  = (unsigned short*)alloc((size_t)NB * 32 * 2);
    float*          alphaB  = (float*)alloc((size_t)NB * 2 * 4);
    float*          beff    = (float*)alloc(18 * 4);

    const int T = 256;
    auto blks = [&](int n) { return (n + T - 1) / T; };

    // ---- phase 1: bf16 conversions ----
    cvt_kernel<<<blks(128 * ND), T, 0, stream>>>(pW1, Wbig, 128 * ND);          // rows 0..127
    zero_kernel<<<blks(32 * ND), T, 0, stream>>>(Wbig + (size_t)128 * ND, 32 * ND);
    cvt_kernel<<<blks(512 * 1024), T, 0, stream>>>(cW2, cW2b, 512 * 1024);
    cvt_kernel<<<blks(512 * 1024), T, 0, stream>>>(aW2, aW2b, 512 * 1024);
    cvt_kernel<<<blks(16 * 512), T, 0, stream>>>(cW3, cW3b, 16 * 512);
    cvt_kernel<<<blks(2 * 512), T, 0, stream>>>(aW3, aW3b, 2 * 512);
    cvt_kernel<<<blks(4 * 32 * 32), T, 0, stream>>>(pW2, pW2b, 4 * 32 * 32);
    cvt_kernel<<<blks(4 * ND * 32), T, 0, stream>>>(pW3, pW3b, 4 * ND * 32);
    cvtT_kernel<<<blks(1024 * ND), T, 0, stream>>>(cW1, cW1Tb, 1024, ND);
    cvtT_kernel<<<blks(1024 * ND), T, 0, stream>>>(aW1, aW1Tb, 1024, ND);
    protoT_kernel<<<blks(ND * 32), T, 0, stream>>>(prototype, protoTb);

    // ---- phase 2: collapse bias chains ----
    bias_kernel<<<1, 1024, 0, stream>>>(cW2, cb1, cb2, cW3, cb3,
                                        aW2, ab1, ab2, aW3, ab3, beff);

    // ---- phase 3: collapse weight chains: Weff = W3 W2 W1 ----
    auto gemm = [&](const unsigned short* A, int lda, const unsigned short* W, int ldw,
                    int M, int N, int K, unsigned short* oB, int ldo, int mode) {
        int tiles = ((M + 15) / 16) * (N / 16);
        gemm_bf16_kernel<<<(tiles + 3) / 4, 128, 0, stream>>>(A, lda, W, ldw, M, N, K,
                                                              oB, ldo, mode);
    };
    gemm(cW2b, 1024, cW1Tb, 1024, 512, ND, 1024, tmp1cT, 512, 2);   // (W2 W1)^T
    gemm(aW2b, 1024, aW1Tb, 1024, 512, ND, 1024, tmp1aT, 512, 2);
    gemm(cW3b, 512, tmp1cT, 512, 16, ND, 512, Wbig + (size_t)128 * ND, ND, 1); // Weff_c
    gemm(aW3b, 512, tmp1aT, 512, 2,  ND, 512, Wbig + (size_t)144 * ND, ND, 1); // Weff_a

    // ---- phase 4: fused first layer of all paths (one pass over feat) ----
    g1_kernel<<<NB / 128, 256, 0, stream>>>(feat, Wbig, pg1, pb1, pm1, pv1, pa1,
                                            t1b, logits);

    // ---- phase 5: softmaxes ----
    softmax_kernel<<<NB / T, T, 0, stream>>>(logits, beff, compab, alphaB);

    // ---- phase 6: trans layer 2 ----
    g2_kernel<<<NB / 16, 128, 0, stream>>>(t1b, pW2b, pg2, pb2, pm2, pv2, pa2, t2b);

    // ---- phase 7: fused layer-3 + attention + alpha blend + residual ----
    final_kernel<<<dim3(ND / 16, NB / 128), 256, 0, stream>>>(
        feat, t2b, pW3b, pg3, pb3, pm3, pv3, pa3, compab, protoTb, alphaB,
        (float*)d_out);
}
